// MoEGateV2_1108101562794
// MI455X (gfx1250) — compile-verified, matrix operand order checked by reading, not run on hardware
//
#include <hip/hip_runtime.h>
#include <math.h>

// Problem sizes (fixed by the reference)
constexpr int Bn = 16, Sn = 4096, Dn = 768;
constexpr int Hn = 12, En = 64, Kn = 8, DHn = 64;
constexpr float SCALE = 0.125f; // 1/sqrt(64)

typedef __attribute__((ext_vector_type(16))) __bf16 v16bf;
typedef __attribute__((ext_vector_type(8)))  __bf16 v8bf;
typedef __attribute__((ext_vector_type(8)))  float  v8f;

// ---------------------------------------------------------------------------
// Kernel A: layernorm(moe_token) -> q = qtok @ Wq^T + bq
//   fold K-projection: rbf[h][k] = bf16( scale * sum_j q[h*64+j]*Wk[h*64+j][k] )
//   (rows 12..15 zero) ; cb[h] = scale * q_h . bk_h
// ---------------------------------------------------------------------------
__global__ void prep_kernel(const float* __restrict__ moe,
                            const float* __restrict__ g,
                            const float* __restrict__ bln,
                            const float* __restrict__ w_in,
                            const float* __restrict__ b_in,
                            __bf16* __restrict__ rbf,
                            float* __restrict__ cb) {
    __shared__ float red[256];
    __shared__ float qtok[Dn];
    __shared__ float qv[Dn];
    __shared__ float stat[2];
    const int tid = threadIdx.x;

    float s = 0.f;
    for (int d = tid; d < Dn; d += 256) s += moe[d];
    red[tid] = s; __syncthreads();
    for (int o = 128; o > 0; o >>= 1) { if (tid < o) red[tid] += red[tid + o]; __syncthreads(); }
    if (tid == 0) stat[0] = red[0] / Dn;
    __syncthreads();
    const float mu = stat[0];

    s = 0.f;
    for (int d = tid; d < Dn; d += 256) { float t = moe[d] - mu; s += t * t; }
    red[tid] = s; __syncthreads();
    for (int o = 128; o > 0; o >>= 1) { if (tid < o) red[tid] += red[tid + o]; __syncthreads(); }
    if (tid == 0) stat[1] = rsqrtf(red[0] / Dn + 1e-6f);
    __syncthreads();
    const float rs = stat[1];

    for (int d = tid; d < Dn; d += 256)
        qtok[d] = (moe[d] - mu) * rs * g[d] + bln[d];
    __syncthreads();

    // q = qtok @ Wq^T + bq   (Wq = rows [0,768) of in_proj_w)
    for (int i = tid; i < Dn; i += 256) {
        float acc = b_in[i];
        const float* w = w_in + (size_t)i * Dn;
        for (int d = 0; d < Dn; ++d) acc += qtok[d] * w[d];
        qv[i] = acc;
    }
    __syncthreads();

    // fold K-projection, store bf16, heads 12..15 zero-padded
    for (int idx = tid; idx < 16 * Dn; idx += 256) {
        const int h = idx / Dn, k = idx % Dn;
        float acc = 0.f;
        if (h < Hn) {
            for (int j = 0; j < DHn; ++j)
                acc += qv[h * DHn + j] * w_in[(size_t)(Dn + h * DHn + j) * Dn + k];
            acc *= SCALE;
        }
        rbf[idx] = (__bf16)acc;
    }
    if (tid < 16) {
        float acc = 0.f;
        if (tid < Hn)
            for (int j = 0; j < DHn; ++j)
                acc += qv[tid * DHn + j] * b_in[Dn + tid * DHn + j];
        cb[tid] = acc * SCALE;
    }
}

// ---------------------------------------------------------------------------
// Kernel B: scores[b][h16][s] = hidden[b,s,:] . r[h,:] + cb[h]
// One wave handles FOUR 16-row S tiles: the bf16 B fragment (r) is loaded once
// per k-step and reused by 4 WMMAs; only streamed hidden needs fp32->bf16.
// ---------------------------------------------------------------------------
__global__ void scores_kernel(const float* __restrict__ hidden,
                              const __bf16* __restrict__ rbf,
                              const float* __restrict__ cb,
                              float* __restrict__ scores) {
    const int b      = blockIdx.y;
    const int s_base = blockIdx.x * 64;
    const int lane   = threadIdx.x;
    const int m      = lane & 15;     // A row (s) / B col (head)
    const int half   = lane >> 4;

    const __bf16* brow = rbf + (size_t)m * Dn;
    const float* arow0 = hidden + ((size_t)b * Sn + s_base +  0 + m) * Dn;
    const float* arow1 = hidden + ((size_t)b * Sn + s_base + 16 + m) * Dn;
    const float* arow2 = hidden + ((size_t)b * Sn + s_base + 32 + m) * Dn;
    const float* arow3 = hidden + ((size_t)b * Sn + s_base + 48 + m) * Dn;
    const float* arows[4] = {arow0, arow1, arow2, arow3};

    v8f acc[4] = {{}, {}, {}, {}};
    for (int k0 = 0; k0 < Dn; k0 += 32) {
        const int ka = k0 + half * 8;
        const int kb = ka + 16;
        __builtin_prefetch(arow0 + k0 + 256, 0, 3);  // global_prefetch_b8

        const v8bf blo = *(const v8bf*)(brow + ka);
        const v8bf bhi = *(const v8bf*)(brow + kb);
        const v16bf bm = __builtin_shufflevector(blo, bhi,
            0,1,2,3,4,5,6,7,8,9,10,11,12,13,14,15);

#pragma unroll
        for (int t = 0; t < 4; ++t) {
            const v8f alo = *(const v8f*)(arows[t] + ka);
            const v8f ahi = *(const v8f*)(arows[t] + kb);
            v16bf a;
#pragma unroll
            for (int e = 0; e < 8; ++e) {
                a[e]     = (__bf16)alo[e];
                a[e + 8] = (__bf16)ahi[e];
            }
            acc[t] = __builtin_amdgcn_wmma_f32_16x16x32_bf16(
                false, a, false, bm, (short)0, acc[t], false, false);
        }
    }

    const int h = lane & 15;
    const float bias = cb[h];
#pragma unroll
    for (int t = 0; t < 4; ++t) {
        float* orow = scores + ((size_t)b * 16 + h) * Sn + s_base + t * 16 + 8 * half;
#pragma unroll
        for (int e = 0; e < 8; ++e) orow[e] = acc[t][e] + bias;  // M = e + 8*half
    }
}

// ---------------------------------------------------------------------------
// Kernel C1: softmax over S per (b,h); writes bf16 attn (heads 12..15 zeroed).
// ---------------------------------------------------------------------------
__global__ void softmax_kernel(const float* __restrict__ scores,
                               __bf16* __restrict__ attnbf) {
    __shared__ float red[256];
    const int h = blockIdx.x, b = blockIdx.y, tid = threadIdx.x;
    __bf16* arow = attnbf + ((size_t)b * 16 + h) * Sn;

    if (h >= Hn) {  // zero-fill pad heads so WMMA A pad rows are clean
        for (int i = tid; i < Sn; i += 256) arow[i] = (__bf16)0.0f;
        return;
    }
    const float* row = scores + ((size_t)b * 16 + h) * Sn;

    float m = -3.4e38f;
    for (int i = tid; i < Sn; i += 256) m = fmaxf(m, row[i]);
    red[tid] = m; __syncthreads();
    for (int o = 128; o > 0; o >>= 1) { if (tid < o) red[tid] = fmaxf(red[tid], red[tid + o]); __syncthreads(); }
    m = red[0]; __syncthreads();

    float s = 0.f;
    for (int i = tid; i < Sn; i += 256) s += expf(row[i] - m);
    red[tid] = s; __syncthreads();
    for (int o = 128; o > 0; o >>= 1) { if (tid < o) red[tid] += red[tid + o]; __syncthreads(); }
    s = red[0]; __syncthreads();

    const float inv = 1.0f / s;
    for (int i = tid; i < Sn; i += 256) arow[i] = (__bf16)(expf(row[i] - m) * inv);
}

// ---------------------------------------------------------------------------
// Kernel C2: U[b][h][d] = sum_s attn[b,h,s] * hidden[b,s,d]
// A fragment: direct bf16 loads from attnbf. B fragment: hidden 32x16 fp32
// tile staged cooperatively (4x b128/lane), converted to a bf16 LDS tile,
// then gathered per WMMA layout. Single-wave WGs -> __syncthreads is S_NOP.
// ---------------------------------------------------------------------------
__global__ void ugemm_kernel(const float* __restrict__ hidden,
                             const __bf16* __restrict__ attnbf,
                             float* __restrict__ u) {
    __shared__ __bf16 tile[32 * 16];
    const int b    = blockIdx.y;
    const int d0   = blockIdx.x * 16;
    const int lane = threadIdx.x;
    const int half = lane >> 4;
    const int dlo  = lane & 15;

    const __bf16* arow = attnbf + ((size_t)b * 16 + dlo) * Sn;  // row = head
    const float* hbase = hidden + (size_t)b * Sn * Dn + d0;

    v8f acc = {};
    for (int s0 = 0; s0 < Sn; s0 += 32) {
        // stage hidden[s0+lane, d0..d0+15] -> bf16 LDS tile (row-major s x 16)
        const float* hr = hbase + (size_t)(s0 + lane) * Dn;
        __builtin_prefetch(hr + 32 * Dn, 0, 3);  // next tile, global_prefetch_b8
        const v8f h0 = *(const v8f*)(hr);
        const v8f h1 = *(const v8f*)(hr + 8);
        v16bf tv;
#pragma unroll
        for (int e = 0; e < 8; ++e) {
            tv[e]     = (__bf16)h0[e];
            tv[e + 8] = (__bf16)h1[e];
        }
        *(v16bf*)&tile[lane * 16] = tv;
        __syncthreads();

        // A fragment: attn rows, contiguous bf16 runs
        const v8bf alo = *(const v8bf*)(arow + s0 + half * 8);
        const v8bf ahi = *(const v8bf*)(arow + s0 + 16 + half * 8);
        const v16bf a = __builtin_shufflevector(alo, ahi,
            0,1,2,3,4,5,6,7,8,9,10,11,12,13,14,15);

        // B fragment: gather column dlo from LDS tile per WMMA K layout
        v16bf bm;
#pragma unroll
        for (int e = 0; e < 8; ++e) {
            bm[e]     = tile[(half * 8 + e) * 16 + dlo];
            bm[e + 8] = tile[(16 + half * 8 + e) * 16 + dlo];
        }
        acc = __builtin_amdgcn_wmma_f32_16x16x32_bf16(
            false, a, false, bm, (short)0, acc, false, false);
        __syncthreads();
    }

#pragma unroll
    for (int e = 0; e < 8; ++e) {
        const int head = e + 8 * half;  // C row M
        if (head < Hn)
            u[((size_t)b * Hn + head) * Dn + d0 + dlo] = acc[e];
    }
}

// ---------------------------------------------------------------------------
// Small GEMVs: ctx = Wv @ U + bv ; x = Wout @ ctx + bout ; logits = x @ gate^T
// ---------------------------------------------------------------------------
__global__ void ctx_kernel(const float* __restrict__ w_in,
                           const float* __restrict__ b_in,
                           const float* __restrict__ u,
                           float* __restrict__ ctx) {
    const int gid = blockIdx.x * blockDim.x + threadIdx.x;  // 16*768
    const int b = gid / Dn, i = gid % Dn;
    const float* w = w_in + (size_t)(2 * Dn + i) * Dn;
    const float* ur = u + ((size_t)b * Hn + i / DHn) * Dn;
    float acc = b_in[2 * Dn + i];
    for (int d = 0; d < Dn; ++d) acc += w[d] * ur[d];
    ctx[gid] = acc;
}

__global__ void outproj_kernel(const float* __restrict__ w_out,
                               const float* __restrict__ b_out,
                               const float* __restrict__ ctx,
                               float* __restrict__ x) {
    const int gid = blockIdx.x * blockDim.x + threadIdx.x;  // 16*768
    const int b = gid / Dn, i = gid % Dn;
    const float* w = w_out + (size_t)i * Dn;
    const float* c = ctx + (size_t)b * Dn;
    float acc = b_out[i];
    for (int d = 0; d < Dn; ++d) acc += w[d] * c[d];
    x[gid] = acc;
}

__global__ void gate_kernel(const float* __restrict__ gate_w,
                            const float* __restrict__ x,
                            float* __restrict__ logits) {
    const int gid = blockIdx.x * blockDim.x + threadIdx.x;  // 16*64
    const int b = gid / En, e = gid % En;
    const float* w = gate_w + (size_t)e * Dn;
    const float* xr = x + (size_t)b * Dn;
    float acc = 0.f;
    for (int d = 0; d < Dn; ++d) acc += w[d] * xr[d];
    logits[gid] = acc;
}

// ---------------------------------------------------------------------------
// Kernel E: per-batch softmax over 64 logits, top-8 (sorted desc), renormalize,
// and aux cross-entropy loss. One wave per batch, single block, no atomics.
// Output layout: [0,128) idx-as-float, [128,256) weights, [256] aux loss.
// ---------------------------------------------------------------------------
__global__ void final_kernel(const float* __restrict__ logits,
                             const int* __restrict__ tgt,
                             float* __restrict__ out) {
    __shared__ float ax[Bn];
    const int tid = threadIdx.x;
    const int b = tid >> 5, lane = tid & 31;

    const float l0 = logits[b * En + lane];
    const float l1 = logits[b * En + 32 + lane];

    float m = fmaxf(l0, l1);
#pragma unroll
    for (int o = 16; o > 0; o >>= 1) m = fmaxf(m, __shfl_xor(m, o, 32));

    const float p0 = expf(l0 - m), p1 = expf(l1 - m);
    float s = p0 + p1;
#pragma unroll
    for (int o = 16; o > 0; o >>= 1) s += __shfl_xor(s, o, 32);

    const float lse = m + logf(s);
    const int t = *tgt;
    const float lt = __shfl((t < 32) ? l0 : l1, t & 31, 32);
    const float auxb = lse - lt;

    const float sc0 = p0 / s, sc1 = p1 / s;
    bool r0 = true, r1 = true;
    float wv[Kn]; int wi[Kn]; float wsum = 0.f;
#pragma unroll
    for (int j = 0; j < Kn; ++j) {
        float v = r0 ? sc0 : -1.0f; int id = lane;
        const float v1 = r1 ? sc1 : -1.0f;
        if (v1 > v) { v = v1; id = lane + 32; }
#pragma unroll
        for (int o = 16; o > 0; o >>= 1) {
            const float ov = __shfl_xor(v, o, 32);
            const int   oi = __shfl_xor(id, o, 32);
            if (ov > v || (ov == v && oi < id)) { v = ov; id = oi; }
        }
        wv[j] = v; wi[j] = id; wsum += v;
        if (id == lane)      r0 = false;
        if (id == lane + 32) r1 = false;
    }
    if (lane == 0) {
        const float inv = 1.0f / (wsum + 1e-20f);
        for (int j = 0; j < Kn; ++j) {
            out[b * Kn + j]           = (float)wi[j];
            out[Bn * Kn + b * Kn + j] = wv[j] * inv;
        }
        ax[b] = auxb;
    }
    __syncthreads();
    if (tid == 0) {
        float a = 0.f;
        for (int i = 0; i < Bn; ++i) a += ax[i];
        out[2 * Bn * Kn] = a / Bn;
    }
}

// ---------------------------------------------------------------------------
extern "C" void kernel_launch(void* const* d_in, const int* in_sizes, int n_in,
                              void* d_out, int out_size, void* d_ws, size_t ws_size,
                              hipStream_t stream) {
    const float* hidden = (const float*)d_in[0];
    const float* moe    = (const float*)d_in[1];
    const float* ln_g   = (const float*)d_in[2];
    const float* ln_b   = (const float*)d_in[3];
    const float* w_in   = (const float*)d_in[4];
    const float* b_in   = (const float*)d_in[5];
    const float* w_out  = (const float*)d_in[6];
    const float* b_out  = (const float*)d_in[7];
    const float* gate_w = (const float*)d_in[8];
    const int*   tgt    = (const int*)d_in[9];
    float* out = (float*)d_out;

    // workspace layout (byte offsets, all 256B-aligned)
    char* w = (char*)d_ws;
    __bf16* rbf    = (__bf16*)w;            w += (size_t)16 * Dn * 2;        // 24576
    float*  cb     = (float*)w;             w += 256;
    float*  scores = (float*)w;             w += (size_t)Bn * 16 * Sn * 4;   // 4 MB
    __bf16* attnbf = (__bf16*)w;            w += (size_t)Bn * 16 * Sn * 2;   // 2 MB
    float*  u      = (float*)w;             w += (size_t)Bn * Hn * Dn * 4;
    float*  ctx    = (float*)w;             w += (size_t)Bn * Dn * 4;
    float*  x      = (float*)w;             w += (size_t)Bn * Dn * 4;
    float*  logits = (float*)w;

    prep_kernel<<<1, 256, 0, stream>>>(moe, ln_g, ln_b, w_in, b_in, rbf, cb);
    scores_kernel<<<dim3(Sn / 64, Bn), 32, 0, stream>>>(hidden, rbf, cb, scores);
    softmax_kernel<<<dim3(16, Bn), 256, 0, stream>>>(scores, attnbf);
    ugemm_kernel<<<dim3(Dn / 16, Bn), 32, 0, stream>>>(hidden, attnbf, u);
    ctx_kernel<<<(Bn * Dn) / 256, 256, 0, stream>>>(w_in, b_in, u, ctx);
    outproj_kernel<<<(Bn * Dn) / 256, 256, 0, stream>>>(w_out, b_out, ctx, x);
    gate_kernel<<<(Bn * En) / 256, 256, 0, stream>>>(gate_w, x, logits);
    final_kernel<<<1, Bn * 32, 0, stream>>>(logits, tgt, out);
}
